// AlbedoKPRNN_41549513622101
// MI455X (gfx1250) — compile-verified, matrix operand order checked by reading, not run on hardware
//
#include <hip/hip_runtime.h>
#include <hip/hip_bf16.h>
#include <cmath>

typedef __attribute__((ext_vector_type(16))) _Float16 v16h;
typedef __attribute__((ext_vector_type(8)))  _Float16 v8h;
typedef __attribute__((ext_vector_type(8)))  float    v8f;

#define HH   128
#define HWN  (128*128)
#define HO   108
#define NPO  (108*108)
#define HK   122
#define NPK  (122*122)
#define KS2  441

// 16-bit A/B fragment K index for (lane, element j) per CDNA5 ISA 16x32 layout:
// lanes 0-15:  V0..3 -> K 0..7,  V4..7 -> K 16..23
// lanes 16-31: V0..3 -> K 8..15, V4..7 -> K 24..31
__device__ __forceinline__ int kmap(int l, int j) {
    return ((l >> 4) << 3) + (j < 8 ? j : j + 8);
}

// ---------------------------------------------------------------------------
// Pack f32 OIHW weights into WMMA-A fragment-ready f16, tap-major K:
//   K' = rs*CinP + c  (CinP = Cin padded to mult of 32, rs = r*3+s)
//   Wp[((g*nChunks+kc)*32 + lane)*16 + j] = W[m=g*16+lane%16][K'=kc*32+kmap(lane,j)]
// ---------------------------------------------------------------------------
__global__ void pack_weights(const float* __restrict__ W, _Float16* __restrict__ Wp,
                             int Cout, int Cin, int CinP) {
    const int nChunks = (9 * CinP) >> 5;
    const int nGroups = (Cout + 15) >> 4;
    const int total   = nGroups * nChunks * 512;
    for (int idx = blockIdx.x * blockDim.x + threadIdx.x; idx < total;
         idx += gridDim.x * blockDim.x) {
        const int j  = idx & 15;
        const int l  = (idx >> 4) & 31;
        const int kc = (idx >> 9) % nChunks;
        const int g  = idx / (nChunks * 512);
        const int K  = kc * 32 + kmap(l, j);
        const int rs = K / CinP;
        const int c  = K - rs * CinP;
        const int m  = g * 16 + (l & 15);
        _Float16 v = (_Float16)0.f;
        if (m < Cout && c < Cin)
            v = (_Float16)W[(size_t)m * Cin * 9 + (size_t)c * 9 + rs];
        Wp[idx] = v;
    }
}

__global__ void f32_to_f16_kernel(const float* __restrict__ src,
                                  _Float16* __restrict__ dst, int n) {
    int i = blockIdx.x * blockDim.x + threadIdx.x;
    if (i < n) dst[i] = (_Float16)src[i];
}

__global__ void zero_f16_kernel(_Float16* __restrict__ p, int n) {
    int i = blockIdx.x * blockDim.x + threadIdx.x;
    if (i < n) p[i] = (_Float16)0.f;
}

__global__ void zero_f32_kernel(float* __restrict__ p, int n) {
    int i = blockIdx.x * blockDim.x + threadIdx.x;
    if (i < n) p[i] = 0.f;
}

// ---------------------------------------------------------------------------
// Implicit-GEMM 3x3 conv via v_wmma_f32_16x16x32_f16.
//   Input = channel concat [X0 (C0 ch) | X1 (rest)] NCHW f16, Ctot channels,
//   CinP = Ctot padded to mult of 32; K' = rs*CinP + c (tap uniform per chunk).
//   Block: 128 thr = 4 waves; wave w -> Cout group g = blockIdx.y*4+w (16 ch),
//   block covers 32 pixels (2 N-tiles per wave, A fragment reused for 2 WMMAs).
//   LDS double-buffered, one barrier per K chunk. Gather is branchless:
//   scalar-clamped channel bases, 8 unconditional loads batched under one
//   wait, cndmask zeroing, one ds_store_b128 per thread per chunk.
//   ACT=0: leaky_relu(0.01) -> f16 Yh ;  ACT=1: softplus -> f32 Yf.
// ---------------------------------------------------------------------------
template <int ACT>
__global__ __launch_bounds__(128)
void conv3x3_wmma(const _Float16* __restrict__ X0, int C0,
                  const _Float16* __restrict__ X1, int Ctot, int CinP,
                  const _Float16* __restrict__ Wp,
                  const float* __restrict__ bias,
                  _Float16* __restrict__ Yh, float* __restrict__ Yf,
                  int Cout, int Hin, int Win, int Hout, int Wout, int pad) {
    const int nChunks = (9 * CinP) >> 5;
    const int Npix    = Hout * Wout;
    const int tid     = threadIdx.x;
    const int lane    = tid & 31;
    // wave id is uniform across the wave: force it into an SGPR so all the
    // gather channel math / pointer selects scalarize.
    const int wave    = __builtin_amdgcn_readfirstlane(tid >> 5);
    const int pixBase = blockIdx.x << 5;            // 32 pixels per block
    const int g       = (blockIdx.y << 2) + wave;   // Cout group (16 ch), scalar
    const int plane   = Hin * Win;

    // ---- fill role: this thread gathers pixel pf for klocal = wave*8 + i ----
    const int pf    = lane;                 // consecutive lanes = consecutive x
    const int pgf   = pixBase + pf;
    const int yf    = pgf / Wout;           // one division, hoisted out of K loop
    const int xf    = pgf - yf * Wout;
    const bool inPix = (pgf < Npix);
    const int ybase = yf * Win;
    const int fragf = pf >> 4;              // which 16-pixel N tile
    const int plf   = pf & 15;
    // this wave's 8 values occupy 8 consecutive halves (16B aligned) in LDS:
    //   hi = wave&1 selects lane half, j0 = (wave>=2)*8 selects element octet
    const int ldsOff = (fragf << 9) + (plf << 4) + ((wave & 1) << 8) +
                       ((wave >= 2) ? 8 : 0);

    __shared__ __align__(32) _Float16 Bsh[2048];   // 2 bufs x 2 frags x 512 halves

    v8f acc0 = {}, acc1 = {};
    int rs = 0, cbase = 0;                  // K' = rs*CinP + cbase + klocal
    for (int kc = 0; kc < nChunks; ++kc) {
        const int r  = rs / 3;              // uniform per chunk (SALU)
        const int s  = rs - r * 3;
        const int dy = r - pad, dx = s - pad;
        const int iy = yf + dy, ix = xf + dx;
        const bool pixOK = inPix && ((unsigned)iy < (unsigned)Hin) &&
                                    ((unsigned)ix < (unsigned)Win);
        const int offSafe = pixOK ? (ybase + dy * Win + ix) : 0;  // clamped, in-bounds

        // phase 1: 8 unconditional loads from scalar-clamped channel bases
        // (single basic block -> loads batch under one s_wait_loadcnt)
        v8h vals;
        #pragma unroll
        for (int i = 0; i < 8; ++i) {
            const int c  = cbase + (wave << 3) + i;     // SGPR: wave-uniform
            const int cc = (c < Ctot) ? c : 0;          // scalar clamp (s_cselect)
            const _Float16* src = (cc < C0) ? (X0 + cc * plane)
                                            : (X1 + (cc - C0) * plane);
            vals[i] = src[offSafe];                     // saddr + voffset, always issued
        }
        // phase 2: zero padded channels / out-of-image lanes (cndmask, no branch)
        #pragma unroll
        for (int i = 0; i < 8; ++i) {
            const int c = cbase + (wave << 3) + i;
            if (!(pixOK && c < Ctot)) vals[i] = (_Float16)0.f;
        }
        *(v8h*)(Bsh + ((kc & 1) << 10) + ldsOff) = vals;  // ds_store_b128

        __syncthreads();                                // single barrier (double buffer)

        const _Float16* buf = Bsh + ((kc & 1) << 10);
        const v16h b0 = *(const v16h*)(buf + (lane << 4));
        const v16h b1 = *(const v16h*)(buf + 512 + (lane << 4));
        const _Float16* aBase = Wp + ((size_t)(g * nChunks + kc) << 9);  // scalar
        const v16h a  = *(const v16h*)(aBase + (lane << 4));
        if (kc + 1 < nChunks)   // global_prefetch_b8 next A fragment
            __builtin_prefetch(aBase + 512 + (lane << 4), 0, 0);

        acc0 = __builtin_amdgcn_wmma_f32_16x16x32_f16(false, a, false, b0,
                                                      (short)0, acc0, false, false);
        acc1 = __builtin_amdgcn_wmma_f32_16x16x32_f16(false, a, false, b1,
                                                      (short)0, acc1, false, false);
        cbase += 32;
        if (cbase >= CinP) { cbase = 0; ++rs; }
    }

    // store: D 16x16 f32 layout -> VGPR v holds M = v + 8*(lane>=16), N = lane%16
    const int mAdd = (lane >> 4) << 3;
    const int pl   = lane & 15;
    const int p0   = pixBase + pl;
    const int p1   = p0 + 16;
    #pragma unroll
    for (int v = 0; v < 8; ++v) {
        const int cout = (g << 4) + v + mAdd;
        if (cout < Cout) {
            const float bv = bias[cout];
            float v0 = acc0[v] + bv;
            float v1 = acc1[v] + bv;
            if (ACT == 0) {
                v0 = v0 > 0.f ? v0 : 0.01f * v0;            // leaky_relu
                v1 = v1 > 0.f ? v1 : 0.01f * v1;
                if (p0 < Npix) Yh[(size_t)cout * Npix + p0] = (_Float16)v0;
                if (p1 < Npix) Yh[(size_t)cout * Npix + p1] = (_Float16)v1;
            } else {
                v0 = v0 > 20.f ? v0 : log1pf(expf(v0));     // softplus
                v1 = v1 > 20.f ? v1 : log1pf(expf(v1));
                if (p0 < Npix) Yf[(size_t)cout * Npix + p0] = v0;
                if (p1 < Npix) Yf[(size_t)cout * Npix + p1] = v1;
            }
        }
    }
}

// ---------------------------------------------------------------------------
// Apply predicted 21x21 kernels (cropped by 7) to irradiance & radiance,
// accumulating weighted sums and |kernel| sums across spp steps.
//   kern: [882][122][122] f32, irr/rad: [3][128][128] f32
// ---------------------------------------------------------------------------
__global__ void apply_kernels_k(const float* __restrict__ kern,
                                const float* __restrict__ irr,
                                const float* __restrict__ rad,
                                float* __restrict__ out_i, float* __restrict__ out_r,
                                float* __restrict__ sw_i,  float* __restrict__ sw_r) {
    const int p = blockIdx.x * blockDim.x + threadIdx.x;
    if (p >= NPO) return;
    const int y = p / HO, x = p - y * HO;
    const float* kp = kern + (size_t)(y + 7) * HK + (x + 7);
    float oi0 = 0, oi1 = 0, oi2 = 0, or0 = 0, or1 = 0, or2 = 0, si = 0, sr = 0;
    for (int ty = 0; ty < 21; ++ty) {
        const size_t row = (size_t)(y + ty) * HH + x;
        const size_t kro = (size_t)ty * 21 * NPK;
        #pragma unroll 3
        for (int tx = 0; tx < 21; ++tx) {
            const float ki = kp[kro + (size_t)tx * NPK];
            const float kr = kp[kro + (size_t)(tx + KS2) * NPK];
            const size_t o = row + tx;
            oi0 += ki * irr[o];
            oi1 += ki * irr[o + HWN];
            oi2 += ki * irr[o + 2 * HWN];
            or0 += kr * rad[o];
            or1 += kr * rad[o + HWN];
            or2 += kr * rad[o + 2 * HWN];
            si += fabsf(ki);
            sr += fabsf(kr);
        }
    }
    out_i[p]           += oi0;
    out_i[p + NPO]     += oi1;
    out_i[p + 2 * NPO] += oi2;
    out_r[p]           += or0;
    out_r[p + NPO]     += or1;
    out_r[p + 2 * NPO] += or2;
    sw_i[p] += si;
    sw_r[p] += sr;
}

// recons = (out_i * mean_spp(albedo)[crop10] + out_r) / (sw_i + sw_r + eps)
__global__ void finalize_k(const float* __restrict__ out_i, const float* __restrict__ out_r,
                           const float* __restrict__ sw_i,  const float* __restrict__ sw_r,
                           const float* __restrict__ albedo, float* __restrict__ out) {
    const int p = blockIdx.x * blockDim.x + threadIdx.x;
    if (p >= NPO) return;
    const int y = p / HO, x = p - y * HO;
    const size_t ap = (size_t)(y + 10) * HH + (x + 10);
    const float inv = 1.0f / (sw_i[p] + sw_r[p] + 1e-8f);
    #pragma unroll
    for (int c = 0; c < 3; ++c) {
        const float alb = 0.5f * (albedo[(size_t)c * HWN + ap] +
                                  albedo[(size_t)(3 + c) * HWN + ap]);
        out[(size_t)c * NPO + p] = (out_i[(size_t)c * NPO + p] * alb +
                                    out_r[(size_t)c * NPO + p]) * inv;
    }
}

// ---------------------------------------------------------------------------
extern "C" void kernel_launch(void* const* d_in, const int* in_sizes, int n_in,
                              void* d_out, int out_size, void* d_ws, size_t ws_size,
                              hipStream_t stream) {
    (void)in_sizes; (void)n_in; (void)out_size; (void)ws_size;
    const float* features   = (const float*)d_in[0];
    const float* radiance   = (const float*)d_in[1];
    const float* irradiance = (const float*)d_in[2];
    const float* albedo     = (const float*)d_in[3];
    const float* W_state    = (const float*)d_in[4];
    const float* b_state    = (const float*)d_in[5];
    const float* W_rnn      = (const float*)d_in[6];
    const float* b_rnn      = (const float*)d_in[7];
    const float* Wk1        = (const float*)d_in[8];
    const float* bk1        = (const float*)d_in[9];
    const float* Wk2        = (const float*)d_in[10];
    const float* bk2        = (const float*)d_in[11];
    const float* Wk3        = (const float*)d_in[12];
    const float* bk3        = (const float*)d_in[13];
    float* out = (float*)d_out;

    // workspace carve (256B aligned)
    char* base = (char*)d_ws;
    size_t off = 0;
    auto carve = [&](size_t bytes) -> void* {
        void* p = base + off;
        off += (bytes + 255) & ~(size_t)255;
        return p;
    };
    auto packHalves = [](int Cout, int CinP) -> size_t {
        return (size_t)((Cout + 15) / 16) * ((9 * CinP) / 32) * 512;
    };

    _Float16* fh   = (_Float16*)carve((size_t)12 * HWN * 2);
    _Float16* stA  = (_Float16*)carve((size_t)64 * HWN * 2);
    _Float16* stB  = (_Float16*)carve((size_t)64 * HWN * 2);
    _Float16* rnn  = (_Float16*)carve((size_t)64 * HWN * 2);
    _Float16* hd1  = (_Float16*)carve((size_t)64 * 126 * 126 * 2);
    _Float16* hd2  = (_Float16*)carve((size_t)64 * 124 * 124 * 2);
    float*    kern = (float*)carve((size_t)(2 * KS2) * NPK * 4);
    float*    oi   = (float*)carve((size_t)3 * NPO * 4);
    float*    orr  = (float*)carve((size_t)3 * NPO * 4);
    float*    si   = (float*)carve((size_t)NPO * 4);
    float*    sr   = (float*)carve((size_t)NPO * 4);
    _Float16* pW1  = (_Float16*)carve(packHalves(64, 96) * 2);
    _Float16* pW2  = (_Float16*)carve(packHalves(64, 64) * 2);
    _Float16* pW3  = (_Float16*)carve(packHalves(64, 96) * 2);
    _Float16* pW4  = (_Float16*)carve(packHalves(64, 64) * 2);
    _Float16* pW5  = (_Float16*)carve(packHalves(2 * KS2, 64) * 2);

    // weight packing f32 -> fragment-ready f16 (tap-major K)
    pack_weights<<<256, 256, 0, stream>>>(W_state, pW1, 64, 76, 96);
    pack_weights<<<256, 256, 0, stream>>>(W_rnn,   pW2, 64, 64, 64);
    pack_weights<<<256, 256, 0, stream>>>(Wk1,     pW3, 64, 76, 96);
    pack_weights<<<256, 256, 0, stream>>>(Wk2,     pW4, 64, 64, 64);
    pack_weights<<<2048, 256, 0, stream>>>(Wk3,    pW5, 2 * KS2, 64, 64);

    // zero initial RNN state + accumulators
    zero_f16_kernel<<<(64 * HWN + 255) / 256, 256, 0, stream>>>(stA, 64 * HWN);
    zero_f32_kernel<<<(3 * NPO + 255) / 256, 256, 0, stream>>>(oi, 3 * NPO);
    zero_f32_kernel<<<(3 * NPO + 255) / 256, 256, 0, stream>>>(orr, 3 * NPO);
    zero_f32_kernel<<<(NPO + 255) / 256, 256, 0, stream>>>(si, NPO);
    zero_f32_kernel<<<(NPO + 255) / 256, 256, 0, stream>>>(sr, NPO);

    _Float16* scur = stA;
    _Float16* snxt = stB;
    for (int s = 0; s < 2; ++s) {
        // features[:, s] -> f16
        f32_to_f16_kernel<<<(12 * HWN + 255) / 256, 256, 0, stream>>>(
            features + (size_t)s * 12 * HWN, fh, 12 * HWN);

        // state = lrelu(conv3x3(concat[f, state]))   SAME, 76->64, 128x128
        conv3x3_wmma<0><<<dim3((HWN + 31) / 32, 1), 128, 0, stream>>>(
            fh, 12, scur, 76, 96, pW1, b_state, snxt, nullptr, 64, HH, HH, HH, HH, 1);
        // rnn = lrelu(conv3x3(state))                SAME, 64->64
        conv3x3_wmma<0><<<dim3((HWN + 31) / 32, 1), 128, 0, stream>>>(
            snxt, 64, nullptr, 64, 64, pW2, b_rnn, rnn, nullptr, 64, HH, HH, HH, HH, 1);
        // hdn1 = lrelu(conv3x3(concat[rnn, f]))      VALID, 76->64, 128->126
        conv3x3_wmma<0><<<dim3((126 * 126 + 31) / 32, 1), 128, 0, stream>>>(
            rnn, 64, fh, 76, 96, pW3, bk1, hd1, nullptr, 64, HH, HH, 126, 126, 0);
        // hdn2 = lrelu(conv3x3(hdn1))                VALID, 64->64, 126->124
        conv3x3_wmma<0><<<dim3((124 * 124 + 31) / 32, 1), 128, 0, stream>>>(
            hd1, 64, nullptr, 64, 64, pW4, bk2, hd2, nullptr, 64, 126, 126, 124, 124, 0);
        // kern = softplus(conv3x3(hdn2))             VALID, 64->882, 124->122
        conv3x3_wmma<1><<<dim3((NPK + 31) / 32, 14), 128, 0, stream>>>(
            hd2, 64, nullptr, 64, 64, pW5, bk3, nullptr, kern, 2 * KS2, 124, 124, HK, HK, 0);

        // weighted 21x21 gather over irradiance/radiance, accumulate across spp
        apply_kernels_k<<<(NPO + 255) / 256, 256, 0, stream>>>(
            kern, irradiance + (size_t)s * 3 * HWN, radiance + (size_t)s * 3 * HWN,
            oi, orr, si, sr);

        _Float16* t = scur; scur = snxt; snxt = t;
    }

    finalize_k<<<(NPO + 255) / 256, 256, 0, stream>>>(oi, orr, si, sr, albedo, out);
}
